// CNNTeacherModel_14551349198854
// MI455X (gfx1250) — matrix operand order repaired and checked by reading, not compiled, not run on hardware
//
#include <hip/hip_runtime.h>
#include <hip/hip_bf16.h>

// ---------------------------------------------------------------------------
// Problem constants (reference: B=32768, H=768, L=10, 3 heads)
// ---------------------------------------------------------------------------
#define HDIM     768
#define NCLS     10
#define NCOLS    32            // 3*10 logits padded to 32 (two 16-wide WMMA tiles)
#define NWAVES   8
#define WGSIZE   (NWAVES * 32)
#define ROWS_PER_WG (NWAVES * 16)
#define KB       32            // K per WMMA
#define KBLOCKS  (HDIM / KB)   // 24
#define KP       776           // padded bf16 LDS row stride (776*2 % 16 == 0)

typedef __attribute__((ext_vector_type(16))) __bf16 v16bf;
typedef __attribute__((ext_vector_type(8)))  __bf16 v8bf;
typedef __attribute__((ext_vector_type(8)))  float  v8f;
typedef __attribute__((ext_vector_type(4)))  float  v4f;

__device__ __forceinline__ v16bf pack_a(v4f a0, v4f a1, v4f a2, v4f a3) {
    v16bf av;
#pragma unroll
    for (int i = 0; i < 4; ++i) {
        av[i]      = (__bf16)a0[i];
        av[4 + i]  = (__bf16)a1[i];
        av[8 + i]  = (__bf16)a2[i];
        av[12 + i] = (__bf16)a3[i];
    }
    return av;
}

__device__ __forceinline__ v16bf pack_b(v8bf lo, v8bf hi) {
    v16bf bv;
#pragma unroll
    for (int i = 0; i < 8; ++i) { bv[i] = lo[i]; bv[8 + i] = hi[i]; }
    return bv;
}

// ---------------------------------------------------------------------------
// Kernel 1: fused 3-head GEMM (bf16 WMMA, f32 accum) + bias + head select +
//           log-softmax cross-entropy; one partial sum per workgroup.
// ---------------------------------------------------------------------------
__global__ __launch_bounds__(WGSIZE, 1)
void ce3head_wmma_kernel(const float* __restrict__ hidden,
                         const float* __restrict__ w1, const float* __restrict__ b1,
                         const float* __restrict__ w2, const float* __restrict__ b2,
                         const float* __restrict__ w3, const float* __restrict__ b3,
                         const int*   __restrict__ groups,
                         const int*   __restrict__ labels,
                         float*       __restrict__ partial,
                         int Btot)
{
    __shared__ __align__(16) __bf16 sW[NCOLS * KP];   // bf16 weights in WMMA-B layout
    __shared__ float sLog[NWAVES][16 * NCOLS];        // per-wave logits tile
    __shared__ float sBias[NCOLS];
    __shared__ float sPart[NWAVES];

    const int tid  = threadIdx.x;
    const int lane = tid & 31;
    const int wave = tid >> 5;

    // ---- Stage all 30 weight rows as bf16 into LDS (once per WG) ----
    for (int idx = tid; idx < NCOLS * HDIM; idx += WGSIZE) {
        const int j = idx / HDIM;
        const int k = idx - j * HDIM;
        float v = 0.0f;
        if      (j < 10) v = w1[j * HDIM + k];
        else if (j < 20) v = w2[(j - 10) * HDIM + k];
        else if (j < 30) v = w3[(j - 20) * HDIM + k];
        sW[j * KP + k] = (__bf16)v;
    }
    if (tid < NCOLS) {
        float bv = 0.0f;
        if      (tid < 10) bv = b1[tid];
        else if (tid < 20) bv = b2[tid - 10];
        else if (tid < 30) bv = b3[tid - 20];
        sBias[tid] = bv;
    }
    __syncthreads();

    // ---- WMMA main loop: each wave computes a 16x32 logits tile ----
    const int rowBase = blockIdx.x * ROWS_PER_WG;
    const int myRowU  = rowBase + wave * 16 + (lane & 15);
    const int myRow   = (myRowU < Btot) ? myRowU : (Btot - 1);

    // A (16-bit 16x32): lane<16 holds K {k0..k0+7, k0+16..k0+23}; lane>=16 same +8
    const int khalfA  = (lane & 16) ? 8 : 0;
    // B (16-bit 32x16): lane<16 holds K {k0..k0+7, k0+16..k0+23}; lane>=16 same +8
    const int khalfB  = (lane & 16) ? 8 : 0;
    const int col     = lane & 15;

    const float*  __restrict__ aptr = hidden + (size_t)myRow * HDIM + khalfA;
    const __bf16* __restrict__ bp0  = &sW[col * KP + khalfB];
    const __bf16* __restrict__ bp1  = &sW[(16 + col) * KP + khalfB];

    v8f acc0 = {};
    v8f acc1 = {};

    // ---- software-pipelined: prologue loads for kb=0 ----
    v4f a0 = *(const v4f*)(aptr + 0);
    v4f a1 = *(const v4f*)(aptr + 4);
    v4f a2 = *(const v4f*)(aptr + 16);
    v4f a3 = *(const v4f*)(aptr + 20);
    v8bf b0lo = *(const v8bf*)(bp0 + 0);
    v8bf b0hi = *(const v8bf*)(bp0 + 16);
    v8bf b1lo = *(const v8bf*)(bp1 + 0);
    v8bf b1hi = *(const v8bf*)(bp1 + 16);

    for (int kb = 0; kb < KBLOCKS - 1; ++kb) {
        // pack current fragments (consumes a*/b* regs, frees them for next loads)
        v16bf av  = pack_a(a0, a1, a2, a3);
        v16bf bv0 = pack_b(b0lo, b0hi);
        v16bf bv1 = pack_b(b1lo, b1hi);

        // issue next-iteration loads BEFORE the WMMAs so their waits overlap
        const int kn = (kb + 1) * KB;
        a0 = *(const v4f*)(aptr + kn);
        a1 = *(const v4f*)(aptr + kn + 4);
        a2 = *(const v4f*)(aptr + kn + 16);
        a3 = *(const v4f*)(aptr + kn + 20);
        b0lo = *(const v8bf*)(bp0 + kn);
        b0hi = *(const v8bf*)(bp0 + kn + 16);
        b1lo = *(const v8bf*)(bp1 + kn);
        b1hi = *(const v8bf*)(bp1 + kn + 16);

        // unconditional speculative prefetch ~3 k-blocks ahead (global_prefetch_b8);
        // translation failures past the buffer end are silently dropped.
        __builtin_prefetch(aptr + kn + 3 * KB, 0, 3);

        acc0 = __builtin_amdgcn_wmma_f32_16x16x32_bf16(false, av, false, bv0,
                                                       (short)0, acc0, false, false);
        acc1 = __builtin_amdgcn_wmma_f32_16x16x32_bf16(false, av, false, bv1,
                                                       (short)0, acc1, false, false);
    }
    // epilogue iteration (kb = KBLOCKS-1), no further loads
    {
        v16bf av  = pack_a(a0, a1, a2, a3);
        v16bf bv0 = pack_b(b0lo, b0hi);
        v16bf bv1 = pack_b(b1lo, b1hi);
        acc0 = __builtin_amdgcn_wmma_f32_16x16x32_bf16(false, av, false, bv0,
                                                       (short)0, acc0, false, false);
        acc1 = __builtin_amdgcn_wmma_f32_16x16x32_bf16(false, av, false, bv1,
                                                       (short)0, acc1, false, false);
    }

    // ---- Spill logits tile to LDS (C/D layout: lane<16 rows 0..7, lane>=16 rows 8..15)
    {
        const int mo = (lane >> 4) * 8;
        float* wl = sLog[wave];
#pragma unroll
        for (int r = 0; r < 8; ++r) {
            wl[(mo + r) * NCOLS + col]      = acc0[r];
            wl[(mo + r) * NCOLS + 16 + col] = acc1[r];
        }
    }
    __syncthreads();

    // ---- Fused cross-entropy: lanes 0..15 each handle one row ----
    float nll = 0.0f;
    {
        const int b = rowBase + wave * 16 + (lane & 15);
        if ((lane < 16) && (b < Btot)) {
            const int g    = groups[b];
            const int lab  = labels[b];
            const int base = g * NCLS;
            const float* wl = &sLog[wave][(lane & 15) * NCOLS];

            float mx = -3.402823466e38f;
            float chosen = 0.0f;
#pragma unroll
            for (int l = 0; l < NCLS; ++l) {
                const float v = wl[base + l] + sBias[base + l];
                mx = fmaxf(mx, v);
                chosen = (l == lab) ? v : chosen;
            }
            float se = 0.0f;
#pragma unroll
            for (int l = 0; l < NCLS; ++l) {
                const float v = wl[base + l] + sBias[base + l];
                se += __expf(v - mx);
            }
            nll = mx - chosen + __logf(se);
        }
    }

    // ---- Deterministic reduction: wave shuffle tree, then in-order wave sum ----
#pragma unroll
    for (int off = 16; off >= 1; off >>= 1)
        nll += __shfl_xor(nll, off, 32);
    if (lane == 0) sPart[wave] = nll;
    __syncthreads();
    if (tid == 0) {
        float s = 0.0f;
#pragma unroll
        for (int w = 0; w < NWAVES; ++w) s += sPart[w];
        partial[blockIdx.x] = s;
    }
}

// ---------------------------------------------------------------------------
// Kernel 2: reduce per-WG partials -> mean NLL scalar (deterministic tree)
// ---------------------------------------------------------------------------
__global__ __launch_bounds__(256, 1)
void ce3head_reduce_kernel(const float* __restrict__ partial,
                           float* __restrict__ out, int nParts, float invB)
{
    __shared__ float red[256];
    const int t = threadIdx.x;
    float s = 0.0f;
    for (int i = t; i < nParts; i += 256) s += partial[i];
    red[t] = s;
    __syncthreads();
#pragma unroll
    for (int off = 128; off > 0; off >>= 1) {
        if (t < off) red[t] += red[t + off];
        __syncthreads();
    }
    if (t == 0) out[0] = red[0] * invB;
}

// ---------------------------------------------------------------------------
// Launch
// ---------------------------------------------------------------------------
extern "C" void kernel_launch(void* const* d_in, const int* in_sizes, int n_in,
                              void* d_out, int out_size, void* d_ws, size_t ws_size,
                              hipStream_t stream) {
    const float* hidden = (const float*)d_in[0];
    const float* w1     = (const float*)d_in[1];
    const float* b1     = (const float*)d_in[2];
    const float* w2     = (const float*)d_in[3];
    const float* b2     = (const float*)d_in[4];
    const float* w3     = (const float*)d_in[5];
    const float* b3     = (const float*)d_in[6];
    const int*   groups = (const int*)d_in[7];
    const int*   labels = (const int*)d_in[8];

    const int B  = in_sizes[0] / HDIM;               // 32768
    const int NB = (B + ROWS_PER_WG - 1) / ROWS_PER_WG;

    float* partials = (float*)d_ws;                  // NB floats of scratch
    float* out      = (float*)d_out;

    ce3head_wmma_kernel<<<NB, WGSIZE, 0, stream>>>(
        hidden, w1, b1, w2, b2, w3, b3, groups, labels, partials, B);

    ce3head_reduce_kernel<<<1, 256, 0, stream>>>(partials, out, NB, 1.0f / (float)B);
}
